// Mutual_MyLayer_53558242181521
// MI455X (gfx1250) — compile-verified, matrix operand order checked by reading, not run on hardware
//
#include <hip/hip_runtime.h>

// ---------------------------------------------------------------------------
// MI455X (gfx1250) attention pipeline, all matmuls on v_wmma_f32_16x16x32_bf16
// (wave32 WMMA).
//
// Pipeline (all on `stream`):
//   1. cvt x,y -> bf16 ; transpose-convert kernel W[3][D][OUT] -> Wt[3][OUT][D]
//   2. GEMM: Q[s][o], K[s][o]  (A = Wt rows o, B = x rows s)        bf16 out
//   3. GEMM: Vt[o][s]          (A = y rows s,  B = Wt2 rows o)      bf16 out
//   4. GEMM: scores[q][k] = Q.K^T                                   f32  out
//   5. softmax rows (scale folded into exp2), write bf16 P in place
//   6. GEMM: out[q][o] = P.V   (A = Vt rows o, B = P rows q)        f32  out
//
// GEMM tiling: 8 waves / 256 threads, block tile 128(m) x 256(n),
// 64x64 per wave = 16 WMMA per 8 fragment loads per k-step of 32
// (2x the FLOP/byte of a 32x32 wave tile; ~220 VGPRs, no spills expected).
//
// Workspace requirement ~157 MB (f32 scores dominate; 64 MB < 192 MB L2).
// ---------------------------------------------------------------------------

typedef unsigned short ushort_t;
typedef __attribute__((ext_vector_type(16))) __bf16 v16bf;
typedef __attribute__((ext_vector_type(8)))  float  v8f;
typedef __attribute__((ext_vector_type(4)))  float  v4f;
typedef __attribute__((ext_vector_type(4)))  unsigned int v4u;

namespace {
constexpr int B_   = 4;
constexpr int S_   = 2048;
constexpr int D_   = 1024;
constexpr int OUT_ = 1024;
constexpr float SCALE = 0.03125f;           // 1/sqrt(1024)
constexpr float LOG2E = 1.4426950408889634f;
}

__device__ __forceinline__ ushort_t f2bf(float f) {
  // round-to-nearest-even fp32 -> bf16
  unsigned u = __float_as_uint(f);
  unsigned r = u + 0x7fffu + ((u >> 16) & 1u);
  return (ushort_t)(r >> 16);
}

// ---- WMMA fragment load -----------------------------------------------------
// 16-bit A/B fragment (16x32 / 32x16): lane L holds row (L&15); elements 0..7
// are k = 8*(L>>4)+0..7, elements 8..15 are k = 16+8*(L>>4)+0..7.
// Both chunks are contiguous 16B loads from row-major storage.
union Frag32 { v4u u[2]; v16bf v; };

__device__ __forceinline__ v16bf load_frag(const ushort_t* base, int ld,
                                           int row, int col) {
  const v4u* p = reinterpret_cast<const v4u*>(base + (size_t)row * ld + col);
  Frag32 f;
  f.u[0] = p[0];   // k = col .. col+7
  f.u[1] = p[2];   // k = col+16 .. col+23   (+32 bytes)
  return f.v;
}

// ---- D-fragment stores ------------------------------------------------------
// D layout: element r, lane L -> m = r + 8*(L>>4), n = L&15.
// Per lane: 8 contiguous-m values -> one 16B bf16 store or two 16B f32 stores.
__device__ __forceinline__ void store_tile(float* dst, const v8f& a) {
  v4f lo = {a[0], a[1], a[2], a[3]};
  v4f hi = {a[4], a[5], a[6], a[7]};
  *reinterpret_cast<v4f*>(dst)     = lo;
  *reinterpret_cast<v4f*>(dst + 4) = hi;
}

__device__ __forceinline__ void store_tile(ushort_t* dst, const v8f& a) {
  union { ushort_t s[8]; v4u u; } pk;
#pragma unroll
  for (int r = 0; r < 8; ++r) pk.s[r] = f2bf(a[r]);
  *reinterpret_cast<v4u*>(dst) = pk.u;
}

// ---- Generic WMMA GEMM ------------------------------------------------------
// Computes D[m][n] = sum_k A[m][k] * B[n][k]  (both operands row-major,
// contiguous k) and stores D at C[n*ldc + m] (contiguous-m per-lane stores).
// Block tile 128(m) x 256(n); 8 waves as 2(m) x 4(n); 64x64 per wave.
template <typename OutT>
__global__ __launch_bounds__(256)
void wmma_gemm_kernel(const ushort_t* __restrict__ A, int lda, long long strideA,
                      const ushort_t* __restrict__ Bm, int ldb, long long strideB,
                      OutT* __restrict__ C, int ldc, long long strideC,
                      int M, int N, int Kd)
{
  const int b = blockIdx.y;
  A  += (size_t)b * strideA;
  Bm += (size_t)b * strideB;
  C  += (size_t)b * strideC;

  const int tid   = threadIdx.x;
  const int wave  = tid >> 5;
  const int lane  = tid & 31;
  const int l15   = lane & 15;
  const int khalf = (lane >> 4) << 3;      // 0 or 8

  const int nBlockM = M >> 7;              // M / 128
  const int bm = blockIdx.x % nBlockM;
  const int bn = blockIdx.x / nBlockM;

  const int m0 = bm * 128 + (wave & 1) * 64;
  const int n0 = bn * 256 + (wave >> 1) * 64;

  const v8f vzero = {0.f, 0.f, 0.f, 0.f, 0.f, 0.f, 0.f, 0.f};
  v8f acc[4][4];
#pragma unroll
  for (int i = 0; i < 4; ++i)
#pragma unroll
    for (int j = 0; j < 4; ++j) acc[i][j] = vzero;

  for (int k = 0; k < Kd; k += 32) {
    v16bf af[4], bf[4];
#pragma unroll
    for (int i = 0; i < 4; ++i)
      af[i] = load_frag(A, lda, m0 + i * 16 + l15, k + khalf);
#pragma unroll
    for (int j = 0; j < 4; ++j)
      bf[j] = load_frag(Bm, ldb, n0 + j * 16 + l15, k + khalf);

#pragma unroll
    for (int i = 0; i < 4; ++i)
#pragma unroll
      for (int j = 0; j < 4; ++j)
        acc[i][j] = __builtin_amdgcn_wmma_f32_16x16x32_bf16(
            false, af[i], false, bf[j], (short)0, acc[i][j], false, false);
  }

#pragma unroll
  for (int i = 0; i < 4; ++i) {
#pragma unroll
    for (int j = 0; j < 4; ++j) {
      const int n  = n0 + j * 16 + l15;
      const int mb = m0 + i * 16 + khalf;
      store_tile(C + (size_t)n * ldc + mb, acc[i][j]);
    }
  }
}

// ---- Row softmax (scale folded), bf16 P written in place over f32 scores ---
__global__ __launch_bounds__(256)
void softmax_rows_kernel(float* __restrict__ sc)
{
  float* rp = sc + (size_t)blockIdx.x * S_;
  const int tid = threadIdx.x;

  float v[8];
  const v4f* vp = reinterpret_cast<const v4f*>(rp + tid * 8);
  v4f a = vp[0], c = vp[1];
  v[0] = a[0]; v[1] = a[1]; v[2] = a[2]; v[3] = a[3];
  v[4] = c[0]; v[5] = c[1]; v[6] = c[2]; v[7] = c[3];

  float lmax = v[0];
#pragma unroll
  for (int r = 1; r < 8; ++r) lmax = fmaxf(lmax, v[r]);

  __shared__ float red[256];
  red[tid] = lmax;
  __syncthreads();
  for (int off = 128; off > 0; off >>= 1) {
    if (tid < off) red[tid] = fmaxf(red[tid], red[tid + off]);
    __syncthreads();
  }
  const float rmax = red[0];
  __syncthreads();

  float lsum = 0.f;
#pragma unroll
  for (int r = 0; r < 8; ++r) {
    v[r] = exp2f((v[r] - rmax) * (SCALE * LOG2E));   // v_exp_f32
    lsum += v[r];
  }
  red[tid] = lsum;
  __syncthreads();
  for (int off = 128; off > 0; off >>= 1) {
    if (tid < off) red[tid] += red[tid + off];
    __syncthreads();
  }
  const float inv = 1.0f / red[0];

  // in-place: bf16 P occupies the first half of this row's f32 slot;
  // all values are already in registers, so no read-after-write hazard.
  ushort_t* op = reinterpret_cast<ushort_t*>(rp) + tid * 8;
  union { ushort_t s[8]; v4u u; } pk;
#pragma unroll
  for (int r = 0; r < 8; ++r) pk.s[r] = f2bf(v[r] * inv);
  *reinterpret_cast<v4u*>(op) = pk.u;
}

// ---- Prep kernels -----------------------------------------------------------
__global__ __launch_bounds__(256)
void cvt_bf16_kernel(const float* __restrict__ in, ushort_t* __restrict__ out,
                     int n)
{
  int i = blockIdx.x * 256 + threadIdx.x;
  if (i < n) out[i] = f2bf(in[i]);
}

__global__ __launch_bounds__(256)
void transpose_w_kernel(const float* __restrict__ W, ushort_t* __restrict__ Wt)
{
  // W[3][D][OUT] -> Wt[3][OUT][D], bf16
  int i = blockIdx.x * 256 + threadIdx.x;           // 3*D*OUT exact multiple
  int w = i / (D_ * OUT_);
  int r = i - w * (D_ * OUT_);
  int d = r / OUT_;
  int o = r - d * OUT_;
  Wt[(size_t)w * OUT_ * D_ + (size_t)o * D_ + d] = f2bf(W[i]);
}

// ---------------------------------------------------------------------------
extern "C" void kernel_launch(void* const* d_in, const int* in_sizes, int n_in,
                              void* d_out, int out_size, void* d_ws,
                              size_t ws_size, hipStream_t stream)
{
  (void)in_sizes; (void)n_in; (void)out_size; (void)ws_size;

  const float* x = (const float*)d_in[0];
  const float* y = (const float*)d_in[1];
  const float* W = (const float*)d_in[2];
  float* out = (float*)d_out;

  // workspace layout (all regions 16B aligned)
  ushort_t* xbf = (ushort_t*)d_ws;                         // B*S*D
  ushort_t* ybf = xbf + (size_t)B_ * S_ * D_;              // B*S*D
  ushort_t* Wt  = ybf + (size_t)B_ * S_ * D_;              // 3*OUT*D
  ushort_t* Qb  = Wt  + (size_t)3 * OUT_ * D_;             // B*S*OUT
  ushort_t* Kb  = Qb  + (size_t)B_ * S_ * OUT_;            // B*S*OUT
  ushort_t* Vt  = Kb  + (size_t)B_ * S_ * OUT_;            // B*OUT*S
  float*    Sc  = (float*)(Vt + (size_t)B_ * OUT_ * S_);   // B*S*S f32

  const int nx = B_ * S_ * D_;
  cvt_bf16_kernel<<<nx / 256, 256, 0, stream>>>(x, xbf, nx);
  cvt_bf16_kernel<<<nx / 256, 256, 0, stream>>>(y, ybf, nx);
  transpose_w_kernel<<<(3 * D_ * OUT_) / 256, 256, 0, stream>>>(W, Wt);

  // Q[s][o] = x.W0 : A = Wt0 (rows o, ld D), B = xbf (rows s, ld D)
  {
    dim3 g((OUT_ / 128) * (S_ / 256), B_);
    wmma_gemm_kernel<ushort_t><<<g, 256, 0, stream>>>(
        Wt, D_, 0LL, xbf, D_, (long long)S_ * D_,
        Qb, OUT_, (long long)S_ * OUT_, OUT_, S_, D_);
  }
  // K[s][o] = x.W1
  {
    dim3 g((OUT_ / 128) * (S_ / 256), B_);
    wmma_gemm_kernel<ushort_t><<<g, 256, 0, stream>>>(
        Wt + (size_t)OUT_ * D_, D_, 0LL, xbf, D_, (long long)S_ * D_,
        Kb, OUT_, (long long)S_ * OUT_, OUT_, S_, D_);
  }
  // Vt[o][s] = (y.W2)^T : A = ybf (rows s), B = Wt2 (rows o)
  {
    dim3 g((S_ / 128) * (OUT_ / 256), B_);
    wmma_gemm_kernel<ushort_t><<<g, 256, 0, stream>>>(
        ybf, D_, (long long)S_ * D_, Wt + (size_t)2 * OUT_ * D_, D_, 0LL,
        Vt, S_, (long long)OUT_ * S_, S_, OUT_, D_);
  }
  // scores[q][k] = Q.K^T : A = Kb (rows k), B = Qb (rows q), f32 out
  {
    dim3 g((S_ / 128) * (S_ / 256), B_);
    wmma_gemm_kernel<float><<<g, 256, 0, stream>>>(
        Kb, OUT_, (long long)S_ * OUT_, Qb, OUT_, (long long)S_ * OUT_,
        Sc, S_, (long long)S_ * S_, S_, S_, OUT_);
  }
  // softmax rows; writes bf16 P in place (row stride stays 2*S_ ushorts)
  softmax_rows_kernel<<<B_ * S_, 256, 0, stream>>>(Sc);

  // out[q][o] = P.V : A = Vt (rows o, ld S), B = P (rows q, ld 2*S), f32 out
  {
    dim3 g((OUT_ / 128) * (S_ / 256), B_);
    wmma_gemm_kernel<float><<<g, 256, 0, stream>>>(
        Vt, S_, (long long)OUT_ * S_,
        (const ushort_t*)Sc, 2 * S_, (long long)2 * S_ * S_,
        out, OUT_, (long long)S_ * OUT_, OUT_, S_, S_);
  }
}
